// Cross_Encoding_56375740727425
// MI455X (gfx1250) — matrix operand order repaired and checked by reading, not compiled, not run on hardware
//
#include <hip/hip_runtime.h>
#include <hip/hip_bf16.h>

typedef __bf16 bf16_t;
typedef __attribute__((ext_vector_type(16))) __bf16 v16bf;
typedef __attribute__((ext_vector_type(8)))  __bf16 v8bf;
typedef __attribute__((ext_vector_type(8)))  float  v8f;
typedef __attribute__((ext_vector_type(8)))  unsigned short us8;

typedef unsigned int ui32x4 __attribute__((ext_vector_type(4)));
typedef int          i32x8  __attribute__((ext_vector_type(8)));
typedef int          i32x4  __attribute__((ext_vector_type(4)));

#define B_SZ 16
#define L_SZ 2048
#define D_SZ 1024
#define NEG_VAL (-1e9f)

#if __has_builtin(__builtin_amdgcn_tensor_load_to_lds)
#define HAVE_TDM 1
#endif

// ---------------- CDNA5 async copy global->LDS ----------------
// Generic shared-space pointers carry the LDS byte offset in their low 32 bits
// (aperture in [63:32]); truncation yields the DS address for VDST.
__device__ __forceinline__ unsigned lds_addr32(const void* p) {
  return (unsigned)(unsigned long long)p;
}

__device__ __forceinline__ void async_copy_b128(unsigned lds_off, const void* gsrc) {
  asm volatile("global_load_async_to_lds_b128 %0, %1, off"
               :: "v"(lds_off), "v"(gsrc)
               : "memory");
}

__device__ __forceinline__ void wait_async0() {
#if __has_builtin(__builtin_amdgcn_s_wait_asynccnt)
  __builtin_amdgcn_s_wait_asynccnt(0);
#else
  asm volatile("s_wait_asynccnt 0x0" ::: "memory");
#endif
}

__device__ __forceinline__ void wait_tensor0() {
#ifdef HAVE_TDM
#if __has_builtin(__builtin_amdgcn_s_wait_tensorcnt)
  __builtin_amdgcn_s_wait_tensorcnt(0);
#else
  asm volatile("s_wait_tensorcnt 0x0" ::: "memory");
#endif
#endif
}

// ---------------- TDM: 2D tile load global -> padded LDS ----------------
// D# per CDNA5 ISA ch.8: group0 = {count/type, lds_addr, global_addr},
// group1 = {flags/pads, tensor dims, tile dims, dim0 stride}. data_size = 2B.
// This toolchain's builtin is the 6-arg variant:
//   (ui32x4 g0, i32x8 g1, i32x4, i32x4, i32x8, i32 cpol)
__device__ __forceinline__ void tdm_load_tile_2d_bf16(
    unsigned lds_off, const void* gaddr,
    unsigned tensor_d0, unsigned tensor_d1, unsigned d0_stride,
    unsigned tile_d0, unsigned tile_d1,
    unsigned pad_interval_code, unsigned pad_amount_code) {
#ifdef HAVE_TDM
  const unsigned long long ga = (unsigned long long)gaddr;
  ui32x4 g0;
  g0.x = 1u;                                  // count=1 (valid user descriptor)
  g0.y = lds_off;                             // lds_addr (bytes)
  g0.z = (unsigned)ga;                        // global_addr[31:0]
  g0.w = (unsigned)(ga >> 32) | (2u << 30);   // global_addr[56:32] | type=2
  i32x8 g1;
  g1[0] = (int)((1u << 16) |                  // data_size = 1 -> 2 bytes
                (1u << 20) |                  // pad_enable
                (pad_interval_code << 22) |   // DWORDs between pads
                (pad_amount_code << 25));     // pad amount (DWORDs)
  g1[1] = (int)(tensor_d0 << 16);             // tensor_dim0[15:0] -> bits 63:48
  g1[2] = (int)((tensor_d0 >> 16) | (tensor_d1 << 16));
  g1[3] = (int)((tensor_d1 >> 16) | (tile_d0 << 16));
  g1[4] = (int)tile_d1;                       // tile_dim1 | tile_dim2=0
  g1[5] = (int)d0_stride;                     // tensor_dim0_stride[31:0]
  g1[6] = 0;
  g1[7] = 0;
  i32x4 gz4 = (i32x4)0;
  i32x8 gz8 = (i32x8)0;
  __builtin_amdgcn_tensor_load_to_lds(g0, g1, gz4, gz4, gz8, 0);
#else
  (void)lds_off; (void)gaddr; (void)tensor_d0; (void)tensor_d1;
  (void)d0_stride; (void)tile_d0; (void)tile_d1;
  (void)pad_interval_code; (void)pad_amount_code;
#endif
}

// ---------------- WMMA fragment loads from LDS ----------------
// A fragment (16x32 bf16): lane L<16 -> M=L, K = {0..7, 16..23};
//                          lane L>=16 -> M=L-16, K = {8..15, 24..31}
__device__ __forceinline__ v16bf load_a_frag(const bf16_t* As, int row_stride,
                                             int m, int lane) {
  const int khalf = (lane >> 4) << 3;  // 0 or 8
  const bf16_t* p = As + (size_t)m * row_stride;
  v16bf a;
  *(v8bf*)&a         = *(const v8bf*)(p + khalf);
  *(((v8bf*)&a) + 1) = *(const v8bf*)(p + khalf + 16);
  return a;
}

// B fragment (32x16 bf16): lane L -> N=L%16, K = {0..7, 16..23} (L<16) or {8..15, 24..31} (L>=16)
__device__ __forceinline__ v16bf load_b_frag(const bf16_t* Bs, int row_stride,
                                             int nb, int lane) {
  const int khalf = (lane >> 4) << 3;  // 0 or 8
  const bf16_t* p = Bs + (size_t)khalf * row_stride + nb + (lane & 15);
  v16bf b;
#pragma unroll
  for (int s = 0; s < 8; ++s) {
    b[s]     = p[(size_t)s * row_stride];
    b[8 + s] = p[(size_t)(16 + s) * row_stride];
  }
  return b;
}

// ---------------- fp32 -> bf16 convert ----------------
__global__ __launch_bounds__(256) void convert_bf16_kernel(
    const float* __restrict__ in, bf16_t* __restrict__ out, int n) {
  int i = blockIdx.x * blockDim.x + threadIdx.x;
  const int stride = gridDim.x * blockDim.x;
  for (; i < n; i += stride) out[i] = (bf16_t)in[i];
}

// ---------------- projection GEMM: Y = bf16(X @ W + bias) ----------------
// X: [M,K] bf16 row-major, W: [K,N] bf16 row-major, Y: [M,N] bf16
// A tile staged via async b128; W tile staged via Tensor Data Mover with
// LDS padding (128 data + 8 pad elements per row -> stride 136).
__global__ __launch_bounds__(256) void proj_gemm_kernel(
    const bf16_t* __restrict__ X, const bf16_t* __restrict__ W,
    const float* __restrict__ bias, bf16_t* __restrict__ Y,
    int M, int K, int N) {
  __shared__ bf16_t As[2][128][40];
  __shared__ bf16_t Bs[2][32][136];

  const int t    = threadIdx.x;
  const int lane = t & 31;
  const int wid  = t >> 5;
  const int wm   = (wid & 3) * 32;
  const int wn   = (wid >> 2) * 64;
  const int m0   = blockIdx.y * 128;
  const int n0   = blockIdx.x * 128;

  v8f acc[2][4] = {};

  const int ar = t >> 1;         // 0..127
  const int ac = (t & 1) * 16;   // 0 or 16
#ifndef HAVE_TDM
  const int br = t >> 3;         // 0..31
  const int bc = (t & 7) * 16;   // 0..112
#endif

  auto stageA = [&](int p, int k0) {
    const bf16_t* xs = X + (size_t)(m0 + ar) * K + k0 + ac;
    async_copy_b128(lds_addr32(&As[p][ar][ac]),     xs);
    async_copy_b128(lds_addr32(&As[p][ar][ac + 8]), xs + 8);
  };
  auto stageB = [&](int p, int k0) {
#ifdef HAVE_TDM
    if (t < 32) {
      // 32x128 bf16 tile; pad every 64 DWORDs (code 5) by 4 DWORDs (code 3)
      tdm_load_tile_2d_bf16(lds_addr32(&Bs[p][0][0]),
                            W + (size_t)k0 * N + n0,
                            (unsigned)N, (unsigned)K, (unsigned)N,
                            128u, 32u, 5u, 3u);
    }
#else
    const bf16_t* ws = W + (size_t)(k0 + br) * N + n0 + bc;
    async_copy_b128(lds_addr32(&Bs[p][br][bc]),     ws);
    async_copy_b128(lds_addr32(&Bs[p][br][bc + 8]), ws + 8);
#endif
  };

  stageA(0, 0);
  stageB(0, 0);
  int p = 0;
  for (int k0 = 0; k0 < K; k0 += 32, p ^= 1) {
    wait_async0();
    wait_tensor0();
    __syncthreads();
    if (k0 + 32 < K) {
      stageA(p ^ 1, k0 + 32);
      stageB(p ^ 1, k0 + 32);
    }

    v16bf afrag[2], bfrag[4];
#pragma unroll
    for (int i = 0; i < 2; ++i)
      afrag[i] = load_a_frag(&As[p][0][0], 40, wm + i * 16 + (lane & 15), lane);
#pragma unroll
    for (int j = 0; j < 4; ++j)
      bfrag[j] = load_b_frag(&Bs[p][0][0], 136, wn + j * 16, lane);
#pragma unroll
    for (int i = 0; i < 2; ++i)
#pragma unroll
      for (int j = 0; j < 4; ++j)
        acc[i][j] = __builtin_amdgcn_wmma_f32_16x16x32_bf16(
            false, afrag[i], false, bfrag[j], (short)0, acc[i][j], false, false);
  }

  const int rbase = m0 + wm + ((lane >> 4) << 3);
  const int cbase = n0 + wn + (lane & 15);
#pragma unroll
  for (int i = 0; i < 2; ++i) {
#pragma unroll
    for (int j = 0; j < 4; ++j) {
      const int col = cbase + j * 16;
      const float bv = bias[col];
#pragma unroll
      for (int r = 0; r < 8; ++r) {
        const int row = rbase + i * 16 + r;
        Y[(size_t)row * N + col] = (bf16_t)(acc[i][j][r] + bv);
      }
    }
  }
}

// ---------------- scores: S[b] = scale * query[b] @ key[b]^T, masked ----------------
__global__ __launch_bounds__(256) void scores_kernel(
    const bf16_t* __restrict__ Qr, const bf16_t* __restrict__ Kr,
    const int* __restrict__ mask, float* __restrict__ S) {
  __shared__ bf16_t As[2][128][40];
  __shared__ bf16_t Bs[2][32][136];

  const int t    = threadIdx.x;
  const int lane = t & 31;
  const int wid  = t >> 5;
  const int wm   = (wid & 3) * 32;
  const int wn   = (wid >> 2) * 64;
  const int b    = blockIdx.z;
  const int m0   = blockIdx.y * 128;  // q
  const int n0   = blockIdx.x * 128;  // a

  const bf16_t* Qb = Qr + (size_t)b * L_SZ * D_SZ;
  const bf16_t* Kb = Kr + (size_t)b * L_SZ * D_SZ;

  v8f acc[2][4] = {};

  const int ar = t >> 1;
  const int ac = (t & 1) * 16;
  const int tn = t >> 1;         // key row index within 128
  const int tk = (t & 1) * 16;   // k sub-chunk

  auto stageA = [&](int p, int k0) {
    const bf16_t* xs = Qb + (size_t)(m0 + ar) * D_SZ + k0 + ac;
    async_copy_b128(lds_addr32(&As[p][ar][ac]),     xs);
    async_copy_b128(lds_addr32(&As[p][ar][ac + 8]), xs + 8);
  };
  // key tile transposed into LDS: Bs[k][n] = key[n0+n][k0+k]
  auto stageB = [&](int p, int k0) {
    const bf16_t* ks = Kb + (size_t)(n0 + tn) * D_SZ + k0 + tk;
    __builtin_prefetch(ks + 32, 0, 3);
    us8 v0 = *(const us8*)(ks);
    us8 v1 = *(const us8*)(ks + 8);
#pragma unroll
    for (int j = 0; j < 8; ++j) {
      Bs[p][tk + j][tn]     = ((const bf16_t*)&v0)[j];
      Bs[p][tk + 8 + j][tn] = ((const bf16_t*)&v1)[j];
    }
  };

  stageA(0, 0);
  stageB(0, 0);
  int p = 0;
  for (int k0 = 0; k0 < D_SZ; k0 += 32, p ^= 1) {
    wait_async0();
    __syncthreads();
    if (k0 + 32 < D_SZ) {
      stageA(p ^ 1, k0 + 32);
      stageB(p ^ 1, k0 + 32);
    }

    v16bf afrag[2], bfrag[4];
#pragma unroll
    for (int i = 0; i < 2; ++i)
      afrag[i] = load_a_frag(&As[p][0][0], 40, wm + i * 16 + (lane & 15), lane);
#pragma unroll
    for (int j = 0; j < 4; ++j)
      bfrag[j] = load_b_frag(&Bs[p][0][0], 136, wn + j * 16, lane);
#pragma unroll
    for (int i = 0; i < 2; ++i)
#pragma unroll
      for (int j = 0; j < 4; ++j)
        acc[i][j] = __builtin_amdgcn_wmma_f32_16x16x32_bf16(
            false, afrag[i], false, bfrag[j], (short)0, acc[i][j], false, false);
  }

  const float scale = 0.03125f;  // 1/sqrt(1024)
  const int*  mb = mask + (size_t)b * L_SZ * L_SZ;
  float*      Sb = S    + (size_t)b * L_SZ * L_SZ;
  const int rbase = m0 + wm + ((lane >> 4) << 3);
  const int cbase = n0 + wn + (lane & 15);
#pragma unroll
  for (int i = 0; i < 2; ++i)
#pragma unroll
    for (int j = 0; j < 4; ++j)
#pragma unroll
      for (int r = 0; r < 8; ++r) {
        const int row = rbase + i * 16 + r;
        const int col = cbase + j * 16;
        const size_t idx = (size_t)row * L_SZ + col;
        float v = acc[i][j][r] * scale;
        if (mb[idx] == 0) v = NEG_VAL;
        Sb[idx] = v;
      }
}

// ---------------- row softmax stats: one wave per row ----------------
__global__ __launch_bounds__(256) void row_stats_kernel(
    const float* __restrict__ S, float* __restrict__ rm, float* __restrict__ rs) {
  const int lane = threadIdx.x & 31;
  const int wid  = threadIdx.x >> 5;
  const int row  = blockIdx.x * 8 + wid;  // 0 .. B*L-1 (S contiguous [B*Lq, La])
  const float* p = S + (size_t)row * L_SZ;
  float m = -3.4e38f;
  for (int i = lane; i < L_SZ; i += 32) m = fmaxf(m, p[i]);
  for (int off = 16; off; off >>= 1) m = fmaxf(m, __shfl_xor(m, off));
  float s = 0.f;
  for (int i = lane; i < L_SZ; i += 32) s += __expf(p[i] - m);
  for (int off = 16; off; off >>= 1) s += __shfl_xor(s, off);
  if (lane == 0) { rm[row] = m; rs[row] = 1.0f / s; }
}

// ---------------- column softmax stats: one thread per column ----------------
__global__ __launch_bounds__(256) void col_stats_kernel(
    const float* __restrict__ S, float* __restrict__ cm, float* __restrict__ cs) {
  const int b = blockIdx.y;
  const int c = blockIdx.x * 256 + threadIdx.x;
  const float* Sb = S + (size_t)b * L_SZ * L_SZ;
  float m = -3.4e38f;
  for (int q = 0; q < L_SZ; ++q) m = fmaxf(m, Sb[(size_t)q * L_SZ + c]);
  float s = 0.f;
  for (int q = 0; q < L_SZ; ++q) s += __expf(Sb[(size_t)q * L_SZ + c] - m);
  cm[(size_t)b * L_SZ + c] = m;
  cs[(size_t)b * L_SZ + c] = 1.0f / s;
}

// ---------------- attn @ value: Out[b] = P @ V[b] ----------------
// TRANS==0: P[i][k] = exp(S[b][i][k]-st_m[i])*st_is[i]   (row softmax, eq)
// TRANS==1: P[i][k] = exp(S[b][k][i]-st_m[i])*st_is[i]   (col softmax, ea)
template <int TRANS>
__global__ __launch_bounds__(256) void attn_av_kernel(
    const float* __restrict__ S, const float* __restrict__ st_m,
    const float* __restrict__ st_is, const bf16_t* __restrict__ V,
    float* __restrict__ Out) {
  __shared__ bf16_t As[2][128][40];
  __shared__ bf16_t Bs[2][32][136];

  const int t    = threadIdx.x;
  const int lane = t & 31;
  const int wid  = t >> 5;
  const int wm   = (wid & 3) * 32;
  const int wn   = (wid >> 2) * 64;
  const int b    = blockIdx.z;
  const int m0   = blockIdx.y * 128;
  const int n0   = blockIdx.x * 128;

  const float*  Sb  = S + (size_t)b * L_SZ * L_SZ;
  const bf16_t* Vb  = V + (size_t)b * L_SZ * D_SZ;
  float*        Ob  = Out + (size_t)b * L_SZ * D_SZ;
  const float*  mB  = st_m  + (size_t)b * L_SZ;
  const float*  isB = st_is + (size_t)b * L_SZ;

  v8f acc[2][4] = {};

  const int br = t >> 3;
  const int bc = (t & 7) * 16;

  auto stageA = [&](int p, int k0) {
    if (TRANS == 0) {
      const int r = t >> 1, cb = (t & 1) * 16;
      const float mv = mB[m0 + r], iv = isB[m0 + r];
      const float4* src = (const float4*)(Sb + (size_t)(m0 + r) * L_SZ + k0 + cb);
      __builtin_prefetch(src + 8, 0, 3);
      v8bf t0, t1;
#pragma unroll
      for (int v4 = 0; v4 < 2; ++v4) {
        float4 f = src[v4];
        t0[v4 * 4 + 0] = (bf16_t)(__expf(f.x - mv) * iv);
        t0[v4 * 4 + 1] = (bf16_t)(__expf(f.y - mv) * iv);
        t0[v4 * 4 + 2] = (bf16_t)(__expf(f.z - mv) * iv);
        t0[v4 * 4 + 3] = (bf16_t)(__expf(f.w - mv) * iv);
      }
#pragma unroll
      for (int v4 = 0; v4 < 2; ++v4) {
        float4 f = src[2 + v4];
        t1[v4 * 4 + 0] = (bf16_t)(__expf(f.x - mv) * iv);
        t1[v4 * 4 + 1] = (bf16_t)(__expf(f.y - mv) * iv);
        t1[v4 * 4 + 2] = (bf16_t)(__expf(f.z - mv) * iv);
        t1[v4 * 4 + 3] = (bf16_t)(__expf(f.w - mv) * iv);
      }
      *(v8bf*)&As[p][r][cb]     = t0;
      *(v8bf*)&As[p][r][cb + 8] = t1;
    } else {
      const int i = t & 127, jb = (t >> 7) * 16;
      const float mv = mB[m0 + i], iv = isB[m0 + i];
#pragma unroll
      for (int j = 0; j < 16; ++j) {
        const float v = Sb[(size_t)(k0 + jb + j) * L_SZ + m0 + i];
        As[p][i][jb + j] = (bf16_t)(__expf(v - mv) * iv);
      }
    }
  };
  auto stageB = [&](int p, int k0) {
    const bf16_t* vs = Vb + (size_t)(k0 + br) * D_SZ + n0 + bc;
    async_copy_b128(lds_addr32(&Bs[p][br][bc]),     vs);
    async_copy_b128(lds_addr32(&Bs[p][br][bc + 8]), vs + 8);
  };

  stageB(0, 0);
  stageA(0, 0);
  int p = 0;
  for (int k0 = 0; k0 < L_SZ; k0 += 32, p ^= 1) {
    wait_async0();
    __syncthreads();
    if (k0 + 32 < L_SZ) {
      stageB(p ^ 1, k0 + 32);
      stageA(p ^ 1, k0 + 32);
    }

    v16bf afrag[2], bfrag[4];
#pragma unroll
    for (int i = 0; i < 2; ++i)
      afrag[i] = load_a_frag(&As[p][0][0], 40, wm + i * 16 + (lane & 15), lane);
#pragma unroll
    for (int j = 0; j < 4; ++j)
      bfrag[j] = load_b_frag(&Bs[p][0][0], 136, wn + j * 16, lane);
#pragma unroll
    for (int i = 0; i < 2; ++i)
#pragma unroll
      for (int j = 0; j < 4; ++j)
        acc[i][j] = __builtin_amdgcn_wmma_f32_16x16x32_bf16(
            false, afrag[i], false, bfrag[j], (short)0, acc[i][j], false, false);
  }

  const int rbase = m0 + wm + ((lane >> 4) << 3);
  const int cbase = n0 + wn + (lane & 15);
#pragma unroll
  for (int i = 0; i < 2; ++i)
#pragma unroll
    for (int j = 0; j < 4; ++j)
#pragma unroll
      for (int r = 0; r < 8; ++r)
        Ob[(size_t)(rbase + i * 16 + r) * D_SZ + cbase + j * 16] = acc[i][j][r];
}

// ---------------- launch ----------------
extern "C" void kernel_launch(void* const* d_in, const int* in_sizes, int n_in,
                              void* d_out, int out_size, void* d_ws, size_t ws_size,
                              hipStream_t stream) {
  const float* Q    = (const float*)d_in[0];
  const float* A    = (const float*)d_in[1];
  const int*   mask = (const int*)d_in[2];
  const float* W1   = (const float*)d_in[3];
  const float* b1   = (const float*)d_in[4];
  const float* W2   = (const float*)d_in[5];
  const float* b2   = (const float*)d_in[6];
  float* out = (float*)d_out;

  char* wsp = (char*)d_ws;
  size_t off = 0;
  auto alloc = [&](size_t bytes) -> void* {
    void* p = wsp + off;
    off += (bytes + 255) & ~(size_t)255;
    return p;
  };
  const size_t nBLD = (size_t)B_SZ * L_SZ * D_SZ;
  const size_t nDD  = (size_t)D_SZ * D_SZ;
  const size_t nBLL = (size_t)B_SZ * L_SZ * L_SZ;

  bf16_t* Qb  = (bf16_t*)alloc(nBLD * 2);
  bf16_t* Ab  = (bf16_t*)alloc(nBLD * 2);
  bf16_t* W1b = (bf16_t*)alloc(nDD * 2);
  bf16_t* W2b = (bf16_t*)alloc(nDD * 2);
  bf16_t* Qrb = (bf16_t*)alloc(nBLD * 2);  // query = Q@W1+b1
  bf16_t* Krb = (bf16_t*)alloc(nBLD * 2);  // key   = A@W2+b2
  float*  S   = (float*)alloc(nBLL * 4);
  float*  rm  = (float*)alloc((size_t)B_SZ * L_SZ * 4);
  float*  rs  = (float*)alloc((size_t)B_SZ * L_SZ * 4);
  float*  cm  = (float*)alloc((size_t)B_SZ * L_SZ * 4);
  float*  cs  = (float*)alloc((size_t)B_SZ * L_SZ * 4);

  convert_bf16_kernel<<<4096, 256, 0, stream>>>(Q, Qb, (int)nBLD);
  convert_bf16_kernel<<<4096, 256, 0, stream>>>(A, Ab, (int)nBLD);
  convert_bf16_kernel<<<512, 256, 0, stream>>>(W1, W1b, (int)nDD);
  convert_bf16_kernel<<<512, 256, 0, stream>>>(W2, W2b, (int)nDD);

  dim3 projGrid(D_SZ / 128, (B_SZ * L_SZ) / 128);
  proj_gemm_kernel<<<projGrid, 256, 0, stream>>>(Qb, W1b, b1, Qrb,
                                                 B_SZ * L_SZ, D_SZ, D_SZ);
  proj_gemm_kernel<<<projGrid, 256, 0, stream>>>(Ab, W2b, b2, Krb,
                                                 B_SZ * L_SZ, D_SZ, D_SZ);

  dim3 scGrid(L_SZ / 128, L_SZ / 128, B_SZ);
  scores_kernel<<<scGrid, 256, 0, stream>>>(Qrb, Krb, mask, S);

  row_stats_kernel<<<(B_SZ * L_SZ) / 8, 256, 0, stream>>>(S, rm, rs);
  col_stats_kernel<<<dim3(L_SZ / 256, B_SZ), 256, 0, stream>>>(S, cm, cs);

  dim3 avGrid(D_SZ / 128, L_SZ / 128, B_SZ);
  attn_av_kernel<0><<<avGrid, 256, 0, stream>>>(S, rm, rs, Ab, out);
  attn_av_kernel<1><<<avGrid, 256, 0, stream>>>(S, cm, cs, Qb,
                                                out + (size_t)B_SZ * L_SZ * D_SZ);
}
